// ScaledDotProductAttention_29291676959364
// MI455X (gfx1250) — matrix operand order repaired and checked by reading, not compiled, not run on hardware
//
#include <hip/hip_runtime.h>
#include <hip/hip_bf16.h>

typedef __attribute__((ext_vector_type(16))) __bf16 v16bf;
typedef __attribute__((ext_vector_type(8)))  float  v8f;
typedef __attribute__((ext_vector_type(4)))  int    v4i;

union BF16Frag { uint4 u[2]; v16bf v; };

#if __has_builtin(__builtin_amdgcn_exp2f)
#define EXP2F(x) __builtin_amdgcn_exp2f(x)
#else
#define EXP2F(x) exp2f(x)
#endif

#define LOG2E 1.44269504088896340736f

#define BATCH 4
#define NLOC  4096          // H*W
#define CDIM  256
#define NROWS (BATCH*NLOC)  // 16384

// ---- CDNA5 async global->LDS copy (ASYNCcnt) with safe fallback ----------
#if __has_builtin(__builtin_amdgcn_global_load_async_to_lds_b128) && \
    __has_builtin(__builtin_amdgcn_s_wait_asynccnt)
#define USE_ASYNC 1
#else
#define USE_ASYNC 0
#endif

static __device__ __forceinline__ void async_cp16(const void* g, void* l) {
#if USE_ASYNC
    // int4 pointers in explicit target address spaces (AS1 = global, AS3 = LDS),
    // constructed via integer round-trip so the attribute binds to the pointee.
    auto* gp = reinterpret_cast<__attribute__((address_space(1))) v4i*>(
        reinterpret_cast<uintptr_t>(g));
    auto* lp = reinterpret_cast<__attribute__((address_space(3))) v4i*>(
        reinterpret_cast<uintptr_t>(l));
    __builtin_amdgcn_global_load_async_to_lds_b128(gp, lp, 0, 0);
#else
    *(uint4*)l = *(const uint4*)g;
#endif
}

static __device__ __forceinline__ void wait_async_le16() {
#if USE_ASYNC
    __builtin_amdgcn_s_wait_asynccnt(16);
#endif
}
static __device__ __forceinline__ void wait_async_le0() {
#if USE_ASYNC
    __builtin_amdgcn_s_wait_asynccnt(0);
#endif
}

static __device__ __forceinline__ v8f wmma_bf16(v16bf a, v16bf b, v8f c) {
    // D = A(16x32 bf16) * B(32x16 bf16) + C(16x16 f32)
    return __builtin_amdgcn_wmma_f32_16x16x32_bf16(
        /*neg_a=*/false, a, /*neg_b=*/false, b,
        /*c_mod=*/(short)0, c, /*reuse_a=*/false, /*reuse_b=*/false);
}

// ---------------------------------------------------------------------------
// Kernel 1: convert x to bf16; convert weights to bf16 transposed (Wt[n][k])
// ---------------------------------------------------------------------------
__global__ void convert_kernel(const float* __restrict__ X,
                               const float* __restrict__ Wv,
                               const float* __restrict__ Wk,
                               const float* __restrict__ Wq,
                               __bf16* __restrict__ xb,
                               __bf16* __restrict__ Wvt,
                               __bf16* __restrict__ Wkt,
                               __bf16* __restrict__ Wqt) {
    int idx = blockIdx.x * blockDim.x + threadIdx.x;
    if (idx < NROWS * CDIM) xb[idx] = (__bf16)X[idx];
    if (idx < CDIM * CDIM) {
        int k = idx / CDIM, n = idx % CDIM;
        Wvt[n * CDIM + k] = (__bf16)Wv[idx];
        Wkt[n * CDIM + k] = (__bf16)Wk[idx];
        Wqt[n * CDIM + k] = (__bf16)Wq[idx];
    }
}

// ---------------------------------------------------------------------------
// Kernel 2: projections. Each wave computes one 16x32 output tile of one of
// v = x@Wv, k = x@Wk, q = x@Wq.  v,k stored row-major bf16; q stored
// transposed per batch: Qt[b][c][n] (so attention B-fragments are contiguous).
// ---------------------------------------------------------------------------
__global__ __launch_bounds__(256) void proj_kernel(
        const __bf16* __restrict__ xb,
        const __bf16* __restrict__ Wvt,
        const __bf16* __restrict__ Wkt,
        const __bf16* __restrict__ Wqt,
        __bf16* __restrict__ Vm,
        __bf16* __restrict__ Km,
        __bf16* __restrict__ Qt) {
    const int lane = threadIdx.x & 31;
    const int wave = threadIdx.x >> 5;
    const int wid  = blockIdx.x * 8 + wave;   // 0 .. 24575
    const int rowTile = wid / 24;             // 0 .. 1023
    const int sub     = wid % 24;
    const int proj    = sub >> 3;             // 0=v 1=k 2=q
    const int ct      = sub & 7;              // column tile (32 cols)
    const int m0 = rowTile * 16;
    const int n0 = ct * 32;
    const int ln = lane & 15;
    const int g  = lane >> 4;

    const __bf16* W = (proj == 0) ? Wvt : (proj == 1) ? Wkt : Wqt;

    v8f acc0 = {0.f,0.f,0.f,0.f,0.f,0.f,0.f,0.f};
    v8f acc1 = acc0;

    const __bf16* xrow  = xb + (size_t)(m0 + ln) * CDIM + g * 16;
    const __bf16* wrow0 = W + (size_t)(n0 + ln) * CDIM + g * 16;
    const __bf16* wrow1 = W + (size_t)(n0 + 16 + ln) * CDIM + g * 16;

#pragma unroll
    for (int c = 0; c < 8; ++c) {
        BF16Frag a;
        a.u[0] = *(const uint4*)(xrow + c * 32);
        a.u[1] = *(const uint4*)(xrow + c * 32 + 8);
        BF16Frag b0, b1;
        b0.u[0] = *(const uint4*)(wrow0 + c * 32);
        b0.u[1] = *(const uint4*)(wrow0 + c * 32 + 8);
        b1.u[0] = *(const uint4*)(wrow1 + c * 32);
        b1.u[1] = *(const uint4*)(wrow1 + c * 32 + 8);
        acc0 = wmma_bf16(a.v, b0.v, acc0);
        acc1 = wmma_bf16(a.v, b1.v, acc1);
    }

#pragma unroll
    for (int t = 0; t < 2; ++t) {
        const v8f acc = t ? acc1 : acc0;
        const int col = n0 + t * 16 + ln;
#pragma unroll
        for (int r = 0; r < 8; ++r) {
            const int row = m0 + r + 8 * g;   // C/D layout: M = r + 8*(lane/16)
            __bf16 h = (__bf16)acc[r];
            if (proj == 0) {
                Vm[(size_t)row * CDIM + col] = h;
            } else if (proj == 1) {
                Km[(size_t)row * CDIM + col] = h;
            } else {
                const int bb = row / NLOC, nl = row % NLOC;
                Qt[((size_t)bb * CDIM + col) * NLOC + nl] = h;
            }
        }
    }
}

// ---------------------------------------------------------------------------
// Kernel 3: fused flash attention (S = v@k^T * gamma, softmax, O = P@q).
// Block = 4 waves = 64 rows. Per 32-col step the k tile (32x256) and q tile
// (256x32) are staged into double-buffered LDS with CDNA5 async global->LDS
// copies (ASYNCcnt), overlapping the next stage with current-step WMMAs.
// ~113 KB static LDS (gfx1250 WGP has 320 KB).
// ---------------------------------------------------------------------------
__global__ __launch_bounds__(128) void attn_kernel(
        const __bf16* __restrict__ Vm,
        const __bf16* __restrict__ Km,
        const __bf16* __restrict__ Qt,
        const float*  __restrict__ X,
        const float*  __restrict__ gamma,
        float* __restrict__ Out) {
    const int tid  = threadIdx.x;
    const int lane = tid & 31;
    const int wave = tid >> 5;
    const int ln = lane & 15;
    const int g  = lane >> 4;
    const int M0 = blockIdx.x * 64;      // block's 64 rows
    const int m0 = M0 + wave * 16;       // this wave's 16 rows
    const int b  = M0 / NLOC;

    // padded LDS (strides chosen for 16B alignment + bank spreading)
    __shared__ __align__(16) __bf16 lds_v[4][16][264];   // per-wave v tiles (A src)
    __shared__ __align__(16) __bf16 ksh[2][32][264];     // double-buffered k tile
    __shared__ __align__(16) __bf16 qsh[2][256][40];     // double-buffered q tile
    __shared__ __align__(16) __bf16 psh[4][16][40];      // per-wave P transpose scratch

    const __bf16* kbase  = Km + (size_t)b * NLOC * CDIM;
    const __bf16* qtbase = Qt + (size_t)b * CDIM * NLOC;

    // issue stage 0 (async) before anything else
    {
#pragma unroll
        for (int i = 0; i < 8; ++i) {
            const int ch = tid + i * 128;
            const int r = ch >> 5, cc = ch & 31;
            async_cp16(kbase + (size_t)r * CDIM + cc * 8, &ksh[0][r][cc * 8]);
        }
#pragma unroll
        for (int i = 0; i < 8; ++i) {
            const int ch = tid + i * 128;
            const int cq = ch >> 2, cc = ch & 3;
            async_cp16(qtbase + (size_t)cq * NLOC + cc * 8, &qsh[0][cq][cc * 8]);
        }
    }

    // load this wave's 16x256 v tile into LDS (wave-private, no barrier needed)
    {
        const __bf16* src = Vm + (size_t)m0 * CDIM;
        for (int i = lane; i < 512; i += 32) {           // 512 chunks of 8 bf16
            const int r = i >> 5, cc = i & 31;
            *(uint4*)(&lds_v[wave][r][cc * 8]) = *(const uint4*)(src + r * CDIM + cc * 8);
        }
    }

    const float gam = gamma[0];

    float mrun[8], lrun[8];
    v8f o[16];
    const v8f vzero = {0.f,0.f,0.f,0.f,0.f,0.f,0.f,0.f};
#pragma unroll
    for (int r = 0; r < 8; ++r) { mrun[r] = -3.0e38f; lrun[r] = 0.f; }
#pragma unroll
    for (int f = 0; f < 16; ++f) o[f] = vzero;

    const int NSTEPS = NLOC / 32;
    for (int it = 0; it < NSTEPS; ++it) {
        const int n0  = it * 32;
        const int cur = it & 1;

        // issue next stage into the other buffer (safe: trailing barrier of the
        // previous iteration guarantees no one still reads it)
        if (it + 1 < NSTEPS) {
            const int nxt = cur ^ 1;
            const int n1 = n0 + 32;
#pragma unroll
            for (int i = 0; i < 8; ++i) {
                const int ch = tid + i * 128;
                const int r = ch >> 5, cc = ch & 31;
                async_cp16(kbase + (size_t)(n1 + r) * CDIM + cc * 8, &ksh[nxt][r][cc * 8]);
            }
#pragma unroll
            for (int i = 0; i < 8; ++i) {
                const int ch = tid + i * 128;
                const int cq = ch >> 2, cc = ch & 3;
                async_cp16(qtbase + (size_t)cq * NLOC + n1 + cc * 8, &qsh[nxt][cq][cc * 8]);
            }
            wait_async_le16();   // retire the 16 older copies (current buffer)
        } else {
            wait_async_le0();
        }
        __syncthreads();         // current buffer fully populated (all waves)

        // ---- S tile: 16 rows x 32 cols, K=256 contraction ----
        v8f s0 = vzero, s1 = vzero;
#pragma unroll
        for (int c = 0; c < 8; ++c) {
            BF16Frag a, k0, k1;
            const __bf16* vp = &lds_v[wave][ln][c * 32 + g * 16];
            a.u[0] = *(const uint4*)(vp);
            a.u[1] = *(const uint4*)(vp + 8);
            const __bf16* kp0 = &ksh[cur][ln][c * 32 + g * 16];
            const __bf16* kp1 = &ksh[cur][16 + ln][c * 32 + g * 16];
            k0.u[0] = *(const uint4*)(kp0);
            k0.u[1] = *(const uint4*)(kp0 + 8);
            k1.u[0] = *(const uint4*)(kp1);
            k1.u[1] = *(const uint4*)(kp1 + 8);
            s0 = wmma_bf16(a.v, k0.v, s0);
            s1 = wmma_bf16(a.v, k1.v, s1);
        }

        // ---- online softmax over this 32-col step ----
#pragma unroll
        for (int r = 0; r < 8; ++r) {
            float t0 = s0[r] * gam;
            float t1 = s1[r] * gam;
            float mx = fmaxf(t0, t1);
#pragma unroll
            for (int off = 1; off < 16; off <<= 1)
                mx = fmaxf(mx, __shfl_xor(mx, off, 16));
            const float mnew = fmaxf(mrun[r], mx);
            const float al = EXP2F((mrun[r] - mnew) * LOG2E);
            const float p0 = EXP2F((t0 - mnew) * LOG2E);
            const float p1 = EXP2F((t1 - mnew) * LOG2E);
            float ps = p0 + p1;
#pragma unroll
            for (int off = 1; off < 16; off <<= 1)
                ps += __shfl_xor(ps, off, 16);
            lrun[r] = lrun[r] * al + ps;
            mrun[r] = mnew;
            // stash P (C/D layout row r+8g, cols ln / 16+ln) for A-layout reload
            psh[wave][r + 8 * g][ln]      = (__bf16)p0;
            psh[wave][r + 8 * g][16 + ln] = (__bf16)p1;
            // rescale running O accumulators (once per 32-col step)
#pragma unroll
            for (int f = 0; f < 16; ++f) o[f][r] *= al;
        }

        // ---- O += P(16x32) @ Q(32x256)  (same-wave DS ops are in-order) ----
        BF16Frag pa;
        {
            const __bf16* pp = &psh[wave][ln][g * 16];
            pa.u[0] = *(const uint4*)(pp);
            pa.u[1] = *(const uint4*)(pp + 8);
        }
#pragma unroll
        for (int f = 0; f < 16; ++f) {
            BF16Frag qb;
            const __bf16* qp = &qsh[cur][f * 16 + ln][g * 16];
            qb.u[0] = *(const uint4*)(qp);
            qb.u[1] = *(const uint4*)(qp + 8);
            o[f] = wmma_bf16(pa.v, qb.v, o[f]);
        }
        __syncthreads();         // all waves done reading cur before it is reused
    }

    // ---- epilogue: normalize, add residual, store f32 ----
#pragma unroll
    for (int f = 0; f < 16; ++f) {
        const int cc = f * 16 + ln;
#pragma unroll
        for (int r = 0; r < 8; ++r) {
            const int row = m0 + r + 8 * g;
            const float val = o[f][r] / lrun[r];
            Out[(size_t)row * CDIM + cc] = val + X[(size_t)row * CDIM + cc];
        }
    }
}

// ---------------------------------------------------------------------------
extern "C" void kernel_launch(void* const* d_in, const int* in_sizes, int n_in,
                              void* d_out, int out_size, void* d_ws, size_t ws_size,
                              hipStream_t stream) {
    const float* X     = (const float*)d_in[0];
    const float* Wv    = (const float*)d_in[1];
    const float* Wk    = (const float*)d_in[2];
    const float* Wq    = (const float*)d_in[3];
    const float* gamma = (const float*)d_in[4];
    float* Out = (float*)d_out;

    // workspace layout (bf16 intermediates), total ~32.4 MB
    char* ws = (char*)d_ws;
    __bf16* xb  = (__bf16*)ws;  ws += (size_t)NROWS * CDIM * 2;        // 8 MB
    __bf16* Vm  = (__bf16*)ws;  ws += (size_t)NROWS * CDIM * 2;        // 8 MB
    __bf16* Km  = (__bf16*)ws;  ws += (size_t)NROWS * CDIM * 2;        // 8 MB
    __bf16* Qt  = (__bf16*)ws;  ws += (size_t)BATCH * CDIM * NLOC * 2; // 8 MB
    __bf16* Wvt = (__bf16*)ws;  ws += (size_t)CDIM * CDIM * 2;
    __bf16* Wkt = (__bf16*)ws;  ws += (size_t)CDIM * CDIM * 2;
    __bf16* Wqt = (__bf16*)ws;  ws += (size_t)CDIM * CDIM * 2;

    // 1) cast x -> bf16, weights -> bf16 transposed
    convert_kernel<<<dim3((NROWS * CDIM) / 256), dim3(256), 0, stream>>>(
        X, Wv, Wk, Wq, xb, Wvt, Wkt, Wqt);

    // 2) projections: 1024 row-tiles x 8 col-tiles x 3 projections = 24576 wave-tiles
    proj_kernel<<<dim3(24576 / 8), dim3(256), 0, stream>>>(
        xb, Wvt, Wkt, Wqt, Vm, Km, Qt);

    // 3) fused flash attention: 256 blocks x 4 waves (64 rows/block)
    attn_kernel<<<dim3(NROWS / 64), dim3(128), 0, stream>>>(
        Vm, Km, Qt, X, gamma, Out);
}
